// DEER_Encoder_652835029630
// MI455X (gfx1250) — matrix operand order repaired and checked by reading, not compiled, not run on hardware
//
#include <hip/hip_runtime.h>
#include <hip/hip_bf16.h>

// ---------------------------------------------------------------------------
// Deformable-DETR encoder for MI455X (gfx1250, wave32).
// GEMM: bf16 WMMA, per-wave 32x64 C tile (8 accumulators), A tile staged in
// LDS via double-buffered global_load_async_to_lds_b128 (ASYNCcnt), B loaded
// directly from L2-resident weights (32B/lane contiguous).
// ---------------------------------------------------------------------------

typedef unsigned short u16b;
typedef unsigned int   u32b;

typedef __attribute__((ext_vector_type(16))) __bf16 v16bf;
typedef __attribute__((ext_vector_type(8)))  __bf16 v8bf;
typedef __attribute__((ext_vector_type(8)))  float  v8f;

#define B_   2
#define D_   256
#define NH_  8
#define DH_  32
#define DFF_ 1024
#define LQ_  12240
#define MT_  24480   /* B_ * LQ_ */

__device__ __forceinline__ u16b f2bf(float f) {
  u32b u = __float_as_uint(f);
  u32b r = (u + 0x7FFFu + ((u >> 16) & 1u)) >> 16;   // round-to-nearest-even
  return (u16b)r;
}

// ------------------------- f32 -> bf16 convert ------------------------------
__global__ void cvt_bf16_kernel(const float* __restrict__ in,
                                u16b* __restrict__ out, size_t n) {
  size_t stride = (size_t)gridDim.x * blockDim.x;
  for (size_t i = (size_t)blockIdx.x * blockDim.x + threadIdx.x; i < n; i += stride)
    out[i] = f2bf(in[i]);
}

// ---------------- x (B,Cin,HW) -> xt (B,HW,Cin) bf16 ------------------------
__global__ void xt_kernel(const float* __restrict__ x, u16b* __restrict__ xt,
                          int Cin, int HW) {
  size_t n = (size_t)B_ * Cin * HW;
  size_t stride = (size_t)gridDim.x * blockDim.x;
  for (size_t i = (size_t)blockIdx.x * blockDim.x + threadIdx.x; i < n; i += stride) {
    int    c  = (int)(i % (size_t)Cin);
    size_t bt = i / (size_t)Cin;
    int    t  = (int)(bt % (size_t)HW);
    int    b  = (int)(bt / (size_t)HW);
    xt[i] = f2bf(x[((size_t)b * Cin + c) * HW + t]);
  }
}

// ------------------------- WMMA bf16 GEMM (tiled) ---------------------------
// C(M,N) = A(M,K) @ W(N,K)^T + bias ; optional relu; f32 and/or bf16 output.
// block = 128 threads (4 waves). Block tile: 32(M) x (4*NSUB*16)(N).
// A tile (32x32 bf16 = 2KB) double-buffered in LDS via async-to-LDS copies.
template<int NSUB>
__global__ void __launch_bounds__(128) wmma_gemm_tiled_kernel(
    const u16b* __restrict__ A, const u16b* __restrict__ Wt,
    const float* __restrict__ bias,
    float* __restrict__ Cf, u16b* __restrict__ Cb,
    int M, int N, int K, int relu)
{
  __shared__ u16b As[2][32 * 32];          // [buf][row*32 + k]

  const int tid  = threadIdx.x;
  const int lane = tid & 31;
  const int wave = tid >> 5;
  const int m0   = blockIdx.x * 32;
  const int n0w  = (blockIdx.y * 4 + wave) * (NSUB * 16);
  const int half = lane >> 4;
  const int l15  = lane & 15;

  // staging decomposition: 128 threads x 16B = 2KB = 32 rows x 64B
  const int srow = tid >> 2;               // 0..31
  const int schk = (tid & 3) * 8;          // bf16 element offset (16B chunks)
  int arow = m0 + srow; if (arow > M - 1) arow = M - 1;   // clamp (pad rows)
  const u16b* Ag = A + (size_t)arow * K + schk;
  const unsigned ldsOff[2] = {
    (unsigned)(size_t)(&As[0][srow * 32 + schk]),
    (unsigned)(size_t)(&As[1][srow * 32 + schk]) };

  const u16b* Brow[NSUB];
#pragma unroll
  for (int nt = 0; nt < NSUB; ++nt)
    Brow[nt] = Wt + (size_t)(n0w + nt * 16 + l15) * K + half * 16;

  v8f zero = {0.f, 0.f, 0.f, 0.f, 0.f, 0.f, 0.f, 0.f};
  v8f acc[2][NSUB];
#pragma unroll
  for (int mt = 0; mt < 2; ++mt)
#pragma unroll
    for (int nt = 0; nt < NSUB; ++nt)
      acc[mt][nt] = zero;

  // prologue: stage k-tile 0 into buffer 0
  asm volatile("global_load_async_to_lds_b128 %0, %1, off"
               :: "v"(ldsOff[0]), "v"(Ag) : "memory");
  asm volatile("s_wait_asynccnt 0x0" ::: "memory");
  __syncthreads();

  const int nIter = K / 32;
  for (int it = 0; it < nIter; ++it) {
    const int cur = it & 1;
    if (it + 1 < nIter) {                  // overlap next A-tile copy
      asm volatile("global_load_async_to_lds_b128 %0, %1, off"
                   :: "v"(ldsOff[cur ^ 1]), "v"(Ag + (size_t)(it + 1) * 32)
                   : "memory");
    }
    // A fragments from LDS (ds_load_b128 x2 per sub-tile)
    v16bf av[2];
#pragma unroll
    for (int mt = 0; mt < 2; ++mt) {
      const u16b* ap = &As[cur][(mt * 16 + l15) * 32 + half * 16];
      v8bf a0 = *(const v8bf*)(ap);
      v8bf a1 = *(const v8bf*)(ap + 8);
      av[mt] = __builtin_shufflevector(a0, a1,
          0,1,2,3,4,5,6,7,8,9,10,11,12,13,14,15);
    }
    const int kg = it * 32;
#pragma unroll
    for (int nt = 0; nt < NSUB; ++nt) {
      v16bf bv = *(const v16bf*)(Brow[nt] + kg);
#pragma unroll
      for (int mt = 0; mt < 2; ++mt)
        acc[mt][nt] = __builtin_amdgcn_wmma_f32_16x16x32_bf16(
            false, av[mt], false, bv, (short)0, acc[mt][nt], false, false);
    }
    if (it + 1 < nIter) {
      asm volatile("s_wait_asynccnt 0x0" ::: "memory");
      __syncthreads();
    }
  }

  // epilogue: stores predicated on M (WMMA already done; divergence OK)
#pragma unroll
  for (int nt = 0; nt < NSUB; ++nt) {
    const int n = n0w + nt * 16 + l15;
    const float bs = bias ? bias[n] : 0.f;
#pragma unroll
    for (int mt = 0; mt < 2; ++mt) {
#pragma unroll
      for (int v = 0; v < 8; ++v) {
        int m = m0 + mt * 16 + v + half * 8;
        if (m < M) {
          float y = acc[mt][nt][v] + bs;
          if (relu) y = fmaxf(y, 0.f);
          size_t idx = (size_t)m * N + n;
          if (Cf) Cf[idx] = y;
          if (Cb) Cb[idx] = f2bf(y);
        }
      }
    }
  }
}

static inline void launch_gemm(const u16b* A, const u16b* Wt, const float* bias,
                               float* Cf, u16b* Cb, int M, int N, int K,
                               int relu, hipStream_t s) {
  int gx = (M + 31) / 32;
  if (N % 256 == 0)
    wmma_gemm_tiled_kernel<4><<<dim3(gx, N / 256), 128, 0, s>>>(
        A, Wt, bias, Cf, Cb, M, N, K, relu);
  else  // N == 128
    wmma_gemm_tiled_kernel<2><<<dim3(gx, 1), 128, 0, s>>>(
        A, Wt, bias, Cf, Cb, M, N, K, relu);
}

// ------------------------- GroupNorm (per batch, level, group) --------------
__global__ void gn_stats_kernel(const float* __restrict__ src,
                                float* __restrict__ stats) {
  const int START[4] = {0, 9216, 11520, 12096};
  const int LS[4]    = {9216, 2304, 576, 144};
  int g = blockIdx.x, lvl = blockIdx.y, b = blockIdx.z;
  int L = LS[lvl], st = START[lvl];
  __shared__ float ss[256], sq[256];
  float s = 0.f, q2 = 0.f;
  for (int i = threadIdx.x; i < L * 8; i += 256) {
    int t = i >> 3, c = g * 8 + (i & 7);
    float v = src[((size_t)(b * LQ_ + st + t)) * 256 + c];
    s += v; q2 += v * v;
  }
  ss[threadIdx.x] = s; sq[threadIdx.x] = q2;
  __syncthreads();
  for (int o = 128; o > 0; o >>= 1) {
    if (threadIdx.x < o) { ss[threadIdx.x] += ss[threadIdx.x + o];
                           sq[threadIdx.x] += sq[threadIdx.x + o]; }
    __syncthreads();
  }
  if (threadIdx.x == 0) {
    float cnt = (float)(L * 8);
    float m = ss[0] / cnt;
    float var = sq[0] / cnt - m * m;
    int sidx = ((b * 4) + lvl) * 32 + g;
    stats[sidx * 2] = m;
    stats[sidx * 2 + 1] = rsqrtf(var + 1e-5f);
  }
}

__global__ void gn_apply_kernel(float* __restrict__ src,
                                const float* __restrict__ stats,
                                const float* __restrict__ g,
                                const float* __restrict__ b) {
  size_t n = (size_t)MT_ * 256;
  size_t stride = (size_t)gridDim.x * blockDim.x;
  for (size_t i = (size_t)blockIdx.x * blockDim.x + threadIdx.x; i < n; i += stride) {
    int c = (int)(i & 255);
    size_t tok = i >> 8;
    int bq = (int)(tok / LQ_);
    int q  = (int)(tok % LQ_);
    int lvl = (q >= 12096) ? 3 : (q >= 11520) ? 2 : (q >= 9216) ? 1 : 0;
    int sidx = ((bq * 4) + lvl) * 32 + (c >> 3);
    float m = stats[sidx * 2], r = stats[sidx * 2 + 1];
    src[i] = (src[i] - m) * r * g[lvl * 256 + c] + b[lvl * 256 + c];
  }
}

// ------------------------- sine pos + level embed + ref points --------------
__global__ void pos_ref_kernel(const float* __restrict__ lvl_emb,
                               float* __restrict__ pos, float* __restrict__ ref) {
  const int START[4] = {0, 9216, 11520, 12096};
  const int HS[4] = {96, 48, 24, 12};
  int i = blockIdx.x * blockDim.x + threadIdx.x;
  if (i >= LQ_ * 256) return;
  int q = i >> 8, d = i & 255;
  int lvl = (q >= 12096) ? 3 : (q >= 11520) ? 2 : (q >= 9216) ? 1 : 0;
  int Hl = HS[lvl], Wl = HS[lvl];
  int t = q - START[lvl];
  int row = t / Wl, col = t % Wl;
  const float scale = 6.28318530717958647692f;
  float e; int idx;
  if (d < 128) { e = (row + 1) / ((float)Hl + 1e-6f) * scale; idx = d; }
  else         { e = (col + 1) / ((float)Wl + 1e-6f) * scale; idx = d - 128; }
  float dimt = __powf(10000.f, (2.f * (float)(idx >> 1)) / 128.f);
  float p = e / dimt;
  float v = (idx & 1) ? __cosf(p) : __sinf(p);
  pos[i] = v + lvl_emb[lvl * 256 + d];
  if (d == 0) {
    ref[q * 2 + 0] = (col + 0.5f) / (float)Wl;
    ref[q * 2 + 1] = (row + 0.5f) / (float)Hl;
  }
}

// ------------------------- q = bf16(src + pos), sb = bf16(src) --------------
__global__ void make_q_kernel(const float* __restrict__ src,
                              const float* __restrict__ pos,
                              u16b* __restrict__ qb, u16b* __restrict__ sb,
                              size_t n, size_t lqd) {
  size_t stride = (size_t)gridDim.x * blockDim.x;
  for (size_t i = (size_t)blockIdx.x * blockDim.x + threadIdx.x; i < n; i += stride) {
    float s = src[i];
    size_t pi = (i >= lqd) ? i - lqd : i;
    qb[i] = f2bf(s + pos[pi]);
    sb[i] = f2bf(s);
  }
}

// ------------------------- softmax over 16 (NL*NP) per head -----------------
__global__ void softmax16_kernel(float* __restrict__ aw, int n) {
  int i = blockIdx.x * blockDim.x + threadIdx.x;
  if (i >= n) return;
  float* p = aw + (size_t)i * 16;
  float mx = p[0];
#pragma unroll
  for (int j = 1; j < 16; ++j) mx = fmaxf(mx, p[j]);
  float e[16], s = 0.f;
#pragma unroll
  for (int j = 0; j < 16; ++j) { e[j] = __expf(p[j] - mx); s += e[j]; }
  float r = 1.f / s;
#pragma unroll
  for (int j = 0; j < 16; ++j) p[j] = e[j] * r;
}

// ------------------------- multi-scale deformable sampling ------------------
__device__ __forceinline__ float samp(const float* __restrict__ vb,
                                      int xx, int yy, int Wl, int Hl, float w) {
  if (xx < 0 || yy < 0 || xx >= Wl || yy >= Hl) return 0.f;
  return vb[(size_t)(yy * Wl + xx) * 256] * w;
}

// one wave per (b, q, head); lane = channel within head (DH_=32).
__global__ void ms_deform_kernel(const float* __restrict__ val,
                                 const float* __restrict__ off,
                                 const float* __restrict__ aw,
                                 const float* __restrict__ ref,
                                 float* __restrict__ attin) {
  const int START[4] = {0, 9216, 11520, 12096};
  const int HS[4] = {96, 48, 24, 12};
  int gw = (blockIdx.x * blockDim.x + threadIdx.x) >> 5;
  int lane = threadIdx.x & 31;
  int h = gw & 7;
  int q = (gw >> 3) % LQ_;
  int b = gw / (8 * LQ_);
  if (b >= B_) return;
  float rx = ref[q * 2 + 0], ry = ref[q * 2 + 1];
  size_t tok = (size_t)b * LQ_ + q;
  const float* offp = off + tok * 256 + h * 32;
  const float* awp  = aw  + tok * 128 + h * 16;
  float acc = 0.f;
#pragma unroll
  for (int l = 0; l < 4; ++l) {
    int Hl = HS[l], Wl = HS[l], st = START[l];
    const float* vb = val + ((size_t)b * LQ_ + st) * 256 + h * 32 + lane;
#pragma unroll
    for (int p = 0; p < 4; ++p) {
      float ox = offp[(l * 4 + p) * 2 + 0];
      float oy = offp[(l * 4 + p) * 2 + 1];
      float x = (rx + ox / (float)Wl) * (float)Wl - 0.5f;
      float y = (ry + oy / (float)Hl) * (float)Hl - 0.5f;
      float x0f = floorf(x), y0f = floorf(y);
      float tx = x - x0f, ty = y - y0f;
      int x0 = (int)x0f, y0 = (int)y0f;
      float w = awp[l * 4 + p];
      acc += samp(vb, x0,     y0,     Wl, Hl, (1.f - tx) * (1.f - ty) * w);
      acc += samp(vb, x0 + 1, y0,     Wl, Hl, tx * (1.f - ty) * w);
      acc += samp(vb, x0,     y0 + 1, Wl, Hl, (1.f - tx) * ty * w);
      acc += samp(vb, x0 + 1, y0 + 1, Wl, Hl, tx * ty * w);
    }
  }
  attin[tok * 256 + h * 32 + lane] = acc;
}

// ------------------------- residual add + LayerNorm -------------------------
__global__ void add_ln_kernel(float* __restrict__ src,
                              const float* __restrict__ add,
                              const float* __restrict__ g,
                              const float* __restrict__ bt,
                              u16b* __restrict__ outb) {
  int tok = blockIdx.x, d = threadIdx.x;
  size_t idx = (size_t)tok * 256 + d;
  float v = src[idx] + add[idx];
  __shared__ float sh[256];
  sh[d] = v; __syncthreads();
  for (int o = 128; o > 0; o >>= 1) { if (d < o) sh[d] += sh[d + o]; __syncthreads(); }
  float m = sh[0] * (1.f / 256.f);
  __syncthreads();
  float dv = v - m;
  sh[d] = dv * dv; __syncthreads();
  for (int o = 128; o > 0; o >>= 1) { if (d < o) sh[d] += sh[d + o]; __syncthreads(); }
  float rinv = rsqrtf(sh[0] * (1.f / 256.f) + 1e-5f);
  float y = dv * rinv * g[d] + bt[d];
  src[idx] = y;
  outb[idx] = f2bf(y);
}

// ------------------------- final output (src + tuple tail) ------------------
__global__ void write_out_kernel(const float* __restrict__ src,
                                 float* __restrict__ out,
                                 size_t nsrc, int out_size) {
  const int LSI[4] = {0, 9216, 11520, 12096};
  const int SHP[8] = {96, 96, 48, 48, 24, 24, 12, 12};
  size_t stride = (size_t)gridDim.x * blockDim.x;
  for (size_t i = (size_t)blockIdx.x * blockDim.x + threadIdx.x;
       i < (size_t)out_size; i += stride) {
    if (i < nsrc) { out[i] = src[i]; continue; }
    int j = (int)(i - nsrc);
    float v;
    if (j < 4)       v = (float)LSI[j];
    else if (j < 12) v = (float)SHP[j - 4];
    else             v = 1.0f;
    out[i] = v;
  }
}

// ===========================================================================
extern "C" void kernel_launch(void* const* d_in, const int* in_sizes, int n_in,
                              void* d_out, int out_size, void* d_ws, size_t ws_size,
                              hipStream_t stream) {
  (void)in_sizes; (void)n_in; (void)ws_size;
  const float* xin[4] = {(const float*)d_in[0], (const float*)d_in[3],
                         (const float*)d_in[6], (const float*)d_in[9]};
  const float* fcw[4] = {(const float*)d_in[1], (const float*)d_in[4],
                         (const float*)d_in[7], (const float*)d_in[10]};
  const float* fcb[4] = {(const float*)d_in[2], (const float*)d_in[5],
                         (const float*)d_in[8], (const float*)d_in[11]};
  const float* gn_g   = (const float*)d_in[12];
  const float* gn_b   = (const float*)d_in[13];
  const float* lvl_e  = (const float*)d_in[14];
  const float* off_w  = (const float*)d_in[15];
  const float* off_b  = (const float*)d_in[16];
  const float* aw_w   = (const float*)d_in[17];
  const float* aw_b   = (const float*)d_in[18];
  const float* val_w  = (const float*)d_in[19];
  const float* val_b  = (const float*)d_in[20];
  const float* out_w  = (const float*)d_in[21];
  const float* out_b  = (const float*)d_in[22];
  const float* ln1_g  = (const float*)d_in[23];
  const float* ln1_b  = (const float*)d_in[24];
  const float* l1_w   = (const float*)d_in[25];
  const float* l1_b   = (const float*)d_in[26];
  const float* l2_w   = (const float*)d_in[27];
  const float* l2_b   = (const float*)d_in[28];
  const float* ln2_g  = (const float*)d_in[29];
  const float* ln2_b  = (const float*)d_in[30];

  // -------- workspace layout (256B aligned blocks) --------
  char* base = (char*)d_ws;
  size_t woff_ = 0;
  auto alloc = [&](size_t bytes) -> char* {
    char* p = base + woff_;
    woff_ += (bytes + 255) & ~(size_t)255;
    return p;
  };
  float* src    = (float*)alloc((size_t)MT_ * 256 * 4);
  float* pos    = (float*)alloc((size_t)LQ_ * 256 * 4);
  float* ref    = (float*)alloc((size_t)LQ_ * 2 * 4);
  u16b*  qb     = (u16b*) alloc((size_t)MT_ * 256 * 2);
  u16b*  sb     = (u16b*) alloc((size_t)MT_ * 256 * 2);
  float* offbuf = (float*)alloc((size_t)MT_ * 256 * 4);
  float* awbuf  = (float*)alloc((size_t)MT_ * 128 * 4);
  float* valbuf = (float*)alloc((size_t)MT_ * 256 * 4);
  float* attin  = (float*)alloc((size_t)MT_ * 256 * 4);
  u16b*  attinb = (u16b*) alloc((size_t)MT_ * 256 * 2);
  float* tmp    = (float*)alloc((size_t)MT_ * 256 * 4);
  u16b*  hb     = (u16b*) alloc((size_t)MT_ * 1024 * 2);
  float* gstat  = (float*)alloc(256 * 2 * 4);
  u16b* wfc  = (u16b*)alloc((size_t)(65536 + 131072 + 196608 + 262144) * 2);
  u16b* woff = (u16b*)alloc((size_t)6 * 65536 * 2);
  u16b* waw  = (u16b*)alloc((size_t)6 * 32768 * 2);
  u16b* wval = (u16b*)alloc((size_t)6 * 65536 * 2);
  u16b* wout = (u16b*)alloc((size_t)6 * 65536 * 2);
  u16b* wl1  = (u16b*)alloc((size_t)6 * 262144 * 2);
  u16b* wl2  = (u16b*)alloc((size_t)6 * 262144 * 2);
  u16b* xt   = (u16b*)alloc((size_t)B_ * 4128768 * 2);

  const int CIN[4]    = {256, 512, 768, 1024};
  const int HW[4]     = {9216, 2304, 576, 144};
  const int START[4]  = {0, 9216, 11520, 12096};
  const size_t FCO[4] = {0, 65536, 196608, 393216};
  size_t XTO[4];
  XTO[0] = 0;
  for (int i = 1; i < 4; ++i)
    XTO[i] = XTO[i - 1] + (size_t)B_ * HW[i - 1] * CIN[i - 1];

  // -------- weight conversion to bf16 --------
  dim3 cg(1024), cb(256);
  for (int i = 0; i < 4; ++i)
    cvt_bf16_kernel<<<cg, cb, 0, stream>>>(fcw[i], wfc + FCO[i], (size_t)256 * CIN[i]);
  cvt_bf16_kernel<<<cg, cb, 0, stream>>>(off_w, woff, (size_t)6 * 65536);
  cvt_bf16_kernel<<<cg, cb, 0, stream>>>(aw_w,  waw,  (size_t)6 * 32768);
  cvt_bf16_kernel<<<cg, cb, 0, stream>>>(val_w, wval, (size_t)6 * 65536);
  cvt_bf16_kernel<<<cg, cb, 0, stream>>>(out_w, wout, (size_t)6 * 65536);
  cvt_bf16_kernel<<<cg, cb, 0, stream>>>(l1_w,  wl1,  (size_t)6 * 262144);
  cvt_bf16_kernel<<<cg, cb, 0, stream>>>(l2_w,  wl2,  (size_t)6 * 262144);

  // -------- input projections (WMMA GEMM) --------
  for (int i = 0; i < 4; ++i)
    xt_kernel<<<cg, cb, 0, stream>>>(xin[i], xt + XTO[i], CIN[i], HW[i]);
  for (int b = 0; b < B_; ++b)
    for (int i = 0; i < 4; ++i)
      launch_gemm(xt + XTO[i] + (size_t)b * HW[i] * CIN[i], wfc + FCO[i], fcb[i],
                  src + ((size_t)b * LQ_ + START[i]) * 256, nullptr,
                  HW[i], 256, CIN[i], 0, stream);

  // -------- group norm + positional embeddings --------
  gn_stats_kernel<<<dim3(32, 4, B_), 256, 0, stream>>>(src, gstat);
  gn_apply_kernel<<<cg, cb, 0, stream>>>(src, gstat, gn_g, gn_b);
  pos_ref_kernel<<<(LQ_ * 256 + 255) / 256, 256, 0, stream>>>(lvl_e, pos, ref);

  // -------- 6 encoder layers --------
  for (int l = 0; l < 6; ++l) {
    make_q_kernel<<<cg, cb, 0, stream>>>(src, pos, qb, sb,
                                         (size_t)MT_ * 256, (size_t)LQ_ * 256);
    launch_gemm(qb, woff + (size_t)l * 65536, off_b + (size_t)l * 256,
                offbuf, nullptr, MT_, 256, 256, 0, stream);
    launch_gemm(qb, waw + (size_t)l * 32768, aw_b + (size_t)l * 128,
                awbuf, nullptr, MT_, 128, 256, 0, stream);
    softmax16_kernel<<<(MT_ * 8 + 255) / 256, 256, 0, stream>>>(awbuf, MT_ * 8);
    launch_gemm(sb, wval + (size_t)l * 65536, val_b + (size_t)l * 256,
                valbuf, nullptr, MT_, 256, 256, 0, stream);
    ms_deform_kernel<<<(MT_ * 8 * 32) / 256, 256, 0, stream>>>(
        valbuf, offbuf, awbuf, ref, attin);
    cvt_bf16_kernel<<<cg, cb, 0, stream>>>(attin, attinb, (size_t)MT_ * 256);
    launch_gemm(attinb, wout + (size_t)l * 65536, out_b + (size_t)l * 256,
                tmp, nullptr, MT_, 256, 256, 0, stream);
    add_ln_kernel<<<MT_, 256, 0, stream>>>(
        src, tmp, ln1_g + (size_t)l * 256, ln1_b + (size_t)l * 256, sb);
    launch_gemm(sb, wl1 + (size_t)l * 262144, l1_b + (size_t)l * 1024,
                nullptr, hb, MT_, 1024, 256, 1, stream);
    launch_gemm(hb, wl2 + (size_t)l * 262144, l2_b + (size_t)l * 256,
                tmp, nullptr, MT_, 256, 1024, 0, stream);
    add_ln_kernel<<<MT_, 256, 0, stream>>>(
        src, tmp, ln2_g + (size_t)l * 256, ln2_b + (size_t)l * 256, sb);
  }

  // -------- output: src then (level_start_index, spatial_shapes, valid_ratios)
  write_out_kernel<<<cg, cb, 0, stream>>>(src, (float*)d_out,
                                          (size_t)MT_ * 256, out_size);
}
